// Attention_78889959293087
// MI455X (gfx1250) — matrix operand order repaired and checked by reading, not compiled, not run on hardware
//
#include <hip/hip_runtime.h>
#include <hip/hip_bf16.h>
#include <math.h>

typedef _Float16 f16;
typedef __attribute__((ext_vector_type(16))) _Float16 v16h;
typedef __attribute__((ext_vector_type(8)))  _Float16 v8h;
typedef __attribute__((ext_vector_type(8)))  float    v8f;
typedef __attribute__((ext_vector_type(4)))  unsigned u32x4;
typedef __attribute__((ext_vector_type(8)))  int      i32x8;

#define LDT 72   // padded LDS row: 64 f16 + 8 pad (144 B, row stride = 36 banks)
#define TILE_BYTES (64 * LDT * 2)

// ---------------------------------------------------------------------------
// Tensor Data Mover: DMA one 64x64 f16 tile (row stride `rs` elements in
// global) into LDS at byte offset `lds_addr`, inserting 16 B of LDS padding
// after every 128 B row (pad_interval = 32 DWORDs -> code 4, pad_amount =
// 4 DWORDs -> code 3), which reproduces the LDT=72 padded layout.
// D# per CDNA5 ISA ch.8: group0 {count, lds_addr, global_addr, type=2},
// group1 {data_size=2B, pad cfg, tensor/tile dims, dim0 stride}.
// Issued by one wave; completion via TENSORcnt (in-order per wave).
// ---------------------------------------------------------------------------
__device__ inline void tdm_load_tile(unsigned lds_addr, const void* gptr,
                                     unsigned rs) {
    unsigned long long ga = (unsigned long long)(size_t)gptr;
    union { u32x4 v; unsigned u[4]; } g0;
    g0.u[0] = 1u;                                         // count=1
    g0.u[1] = lds_addr;                                   // lds_addr [63:32]
    g0.u[2] = (unsigned)ga;                               // global_addr lo
    g0.u[3] = (unsigned)((ga >> 32) & 0x1FFFFFFull)       // global_addr[56:32]
            | (2u << 30);                                 // type=2 (image)
    union { i32x8 v; unsigned u[8]; } g1;
    g1.u[0] = (1u << 16)            // data_size = 2 bytes
            | (1u << 20)            // pad_enable
            | (4u << 22)            // pad_interval: 32 DWORDs
            | (3u << 25);           // pad_amount: 4 DWORDs (16 B)
    g1.u[1] = (rs & 0xFFFFu) << 16;                 // tensor_dim0 [15:0]
    g1.u[2] = (rs >> 16) | (0u << 16);              // tensor_dim0 hi, dim1 lo
    g1.u[3] = 16u | (64u << 16);                    // tensor_dim1 hi (=1<<20), tile_dim0=64
    g1.u[4] = 64u;                                  // tile_dim1=64, tile_dim2=0
    g1.u[5] = rs;                                   // tensor_dim0_stride lo32
    g1.u[6] = 0u;                                   // stride hi, dim1_stride lo
    g1.u[7] = 0u;
    asm volatile("tensor_load_to_lds %0, %1" :: "s"(g0.v), "s"(g1.v) : "memory");
}

__device__ inline unsigned lds_off(const void* p) {
    return (unsigned)(size_t)p;    // generic LDS address: low 32 bits = offset
}

// ---------------------------------------------------------------------------
// WMMA fragment loaders (CDNA5 wave32 16x16x32 f16 layouts, ISA 7.12.2)
// ---------------------------------------------------------------------------
__device__ inline v16h load_a_frag(const f16* rowp, int k0, int half) {
    union { v16h v; v8h h[2]; } u;
    u.h[0] = *(const v8h*)(rowp + k0 + half * 8);
    u.h[1] = *(const v8h*)(rowp + k0 + 16 + half * 8);
    return u.v;
}

__device__ inline v16h load_b_frag(const f16* rowp, int koff, int half) {
    union { v16h v; v8h h[2]; } u;
    const f16* p = rowp + koff + half * 8;
    u.h[0] = *(const v8h*)(p);
    u.h[1] = *(const v8h*)(p + 16);
    return u.v;
}

__device__ inline v8f wmma_f16(v16h a, v16h b, v8f c) {
    return __builtin_amdgcn_wmma_f32_16x16x32_f16(false, a, false, b,
                                                  (short)0, c, false, false);
}

__device__ inline float hmax16(float v) {
    #pragma unroll
    for (int m = 1; m < 16; m <<= 1) v = fmaxf(v, __shfl_xor(v, m, 32));
    return v;
}
__device__ inline float hsum16(float v) {
    #pragma unroll
    for (int m = 1; m < 16; m <<= 1) v += __shfl_xor(v, m, 32);
    return v;
}

// ---------------------------------------------------------------------------
// Kernel 1a: x (f32) -> xh (f16), elementwise
// ---------------------------------------------------------------------------
__global__ __launch_bounds__(256) void convert_x(const float* __restrict__ x,
                                                 f16* __restrict__ xh) {
    int i = blockIdx.x * 256 + threadIdx.x;
    xh[i] = (f16)x[i];
}

// Kernel 1b: W (768 x 2304 f32) -> whT (2304 x 768 f16)
__global__ __launch_bounds__(256) void convert_wT(const float* __restrict__ W,
                                                  f16* __restrict__ whT) {
    int i = blockIdx.x * 256 + threadIdx.x;
    int k = i / 2304;
    int c = i - k * 2304;
    whT[(size_t)c * 768 + k] = (f16)W[i];
}

// ---------------------------------------------------------------------------
// Kernel 2: QKV = X @ W + b  (M=8192, N=2304, K=768), TDM double-buffered.
// Block = 4 waves -> 64x64 C tile; wave w owns rows [16w,16w+16).
// ---------------------------------------------------------------------------
__global__ __launch_bounds__(128) void qkv_gemm(const f16* __restrict__ xh,
                                                const f16* __restrict__ whT,
                                                const float* __restrict__ bias,
                                                f16* __restrict__ qh,
                                                f16* __restrict__ kh,
                                                f16* __restrict__ vth) {
    __shared__ __align__(16) f16 ldsX[2][64 * LDT];
    __shared__ __align__(16) f16 ldsW[2][64 * LDT];

    const int tid  = threadIdx.x;
    const int lane = tid & 31;
    const int wv   = tid >> 5;
    const int half = (lane >> 4) & 1;
    const int l16  = lane & 15;
    const int m0   = blockIdx.x * 64;
    const int c0   = blockIdx.y * 64;

    const f16* xbase = xh  + (size_t)m0 * 768;
    const f16* wbase = whT + (size_t)c0 * 768;

    if (wv == 0) {
        tdm_load_tile(lds_off(&ldsX[0][0]), xbase, 768);
        tdm_load_tile(lds_off(&ldsW[0][0]), wbase, 768);
    }

    v8f acc[4] = {};

    for (int t = 0; t < 12; ++t) {
        const int cur = t & 1;
        if (wv == 0 && t + 1 < 12) {
            tdm_load_tile(lds_off(&ldsX[cur ^ 1][0]), xbase + (t + 1) * 64, 768);
            tdm_load_tile(lds_off(&ldsW[cur ^ 1][0]), wbase + (t + 1) * 64, 768);
        }
        if (t + 1 < 12) __builtin_amdgcn_s_wait_tensorcnt(2);
        else            __builtin_amdgcn_s_wait_tensorcnt(0);
        __syncthreads();

        const f16* arow = &ldsX[cur][(wv * 16 + l16) * LDT];
        v16h a0 = load_a_frag(arow, 0,  half);
        v16h a1 = load_a_frag(arow, 32, half);
        #pragma unroll
        for (int jt = 0; jt < 4; ++jt) {
            const f16* brow = &ldsW[cur][(jt * 16 + l16) * LDT];
            v16h b0 = load_b_frag(brow, 0,  half);
            v16h b1 = load_b_frag(brow, 32, half);
            acc[jt] = wmma_f16(a0, b0, acc[jt]);
            acc[jt] = wmma_f16(a1, b1, acc[jt]);
        }
        __syncthreads();
    }

    // epilogue: bias add + f16 scatter into q[b,h,n,d], k[b,h,n,d], vT[b,h,d,n]
    const int trow0 = m0 + wv * 16 + half * 8;
    #pragma unroll
    for (int jt = 0; jt < 4; ++jt) {
        int c = c0 + jt * 16 + l16;
        float bv = bias[c];
        int s   = c / 768;
        int rem = c - s * 768;
        int h   = rem >> 6;
        int d   = rem & 63;
        #pragma unroll
        for (int r = 0; r < 8; ++r) {
            int t  = trow0 + r;
            int bb = t >> 10;
            int n  = t & 1023;
            f16 hv = (f16)(acc[jt][r] + bv);
            size_t bh = (size_t)bb * 12 + h;
            if (s == 0)      qh [(bh * 1024 + n) * 64 + d] = hv;
            else if (s == 1) kh [(bh * 1024 + n) * 64 + d] = hv;
            else             vth[(bh * 64 + d) * 1024 + n] = hv;
        }
    }
}

// ---------------------------------------------------------------------------
// Kernel 3: flash-style attention per (b,h), 64 query rows per block.
// S = Q K^T (no 1/sqrt(Dh), faithful to reference), online softmax, O = P V.
// K/V^T tiles TDM double-buffered.
// ---------------------------------------------------------------------------
__global__ __launch_bounds__(128) void attn(const f16* __restrict__ qh,
                                            const f16* __restrict__ kh,
                                            const f16* __restrict__ vth,
                                            float* __restrict__ out) {
    __shared__ __align__(16) f16 ldsQ[64 * LDT];
    __shared__ __align__(16) f16 ldsK[2][64 * LDT];
    __shared__ __align__(16) f16 ldsV[2][64 * LDT];    // V^T tile: [d][j]
    __shared__ __align__(16) f16 ldsP[4 * 16 * LDT];   // per-wave P tile

    const int tid  = threadIdx.x;
    const int lane = tid & 31;
    const int wv   = tid >> 5;
    const int half = (lane >> 4) & 1;
    const int l16  = lane & 15;
    const int bh   = blockIdx.x >> 4;
    const int qb   = blockIdx.x & 15;
    const int b    = bh / 12;
    const int h    = bh - b * 12;
    const size_t base = (size_t)bh * 1024 * 64;

    if (wv == 0) {
        tdm_load_tile(lds_off(&ldsQ[0]),    qh  + base + (size_t)qb * 4096, 64);
        tdm_load_tile(lds_off(&ldsK[0][0]), kh  + base,                     64);
        tdm_load_tile(lds_off(&ldsV[0][0]), vth + base,                     1024);
    }
    __builtin_amdgcn_s_wait_tensorcnt(2);   // Q tile landed (K0/V0 may be in flight)
    __syncthreads();

    const f16* qrow = &ldsQ[(wv * 16 + l16) * LDT];
    const v16h qa0 = load_a_frag(qrow, 0,  half);
    const v16h qa1 = load_a_frag(qrow, 32, half);

    v8f o[4] = {};
    float mrow[8], lrow[8];
    #pragma unroll
    for (int r = 0; r < 8; ++r) { mrow[r] = -1e30f; lrow[r] = 0.f; }

    for (int kb = 0; kb < 16; ++kb) {
        const int cur = kb & 1;
        if (wv == 0 && kb + 1 < 16) {
            tdm_load_tile(lds_off(&ldsK[cur ^ 1][0]),
                          kh + base + (size_t)(kb + 1) * 4096, 64);
            tdm_load_tile(lds_off(&ldsV[cur ^ 1][0]),
                          vth + base + (size_t)(kb + 1) * 64, 1024);
        }
        if (kb + 1 < 16) __builtin_amdgcn_s_wait_tensorcnt(2);
        else             __builtin_amdgcn_s_wait_tensorcnt(0);
        __syncthreads();

        // S tiles: 16 rows x 64 cols, K-dim = d (64) -> 2 WMMAs per tile
        v8f S[4];
        #pragma unroll
        for (int jt = 0; jt < 4; ++jt) {
            const f16* krow = &ldsK[cur][(jt * 16 + l16) * LDT];
            v16h b0 = load_b_frag(krow, 0,  half);
            v16h b1 = load_b_frag(krow, 32, half);
            v8f c = {};
            c = wmma_f16(qa0, b0, c);
            c = wmma_f16(qa1, b1, c);
            S[jt] = c;
        }

        // online softmax (C-fragment rows live in 16-lane halves)
        float scale[8];
        #pragma unroll
        for (int r = 0; r < 8; ++r) {
            float mx = fmaxf(fmaxf(S[0][r], S[1][r]), fmaxf(S[2][r], S[3][r]));
            mx = hmax16(mx);
            float mn = fmaxf(mrow[r], mx);
            scale[r] = __expf(mrow[r] - mn);
            mrow[r]  = mn;
        }
        #pragma unroll
        for (int jt = 0; jt < 4; ++jt)
            #pragma unroll
            for (int r = 0; r < 8; ++r)
                S[jt][r] = __expf(S[jt][r] - mrow[r]);
        #pragma unroll
        for (int r = 0; r < 8; ++r) {
            float s = hsum16(S[0][r] + S[1][r] + S[2][r] + S[3][r]);
            lrow[r] = lrow[r] * scale[r] + s;
        }

        // P -> LDS (f16 row-major) so it can be reloaded in A-operand layout
        f16* pbase = &ldsP[wv * 16 * LDT];
        #pragma unroll
        for (int jt = 0; jt < 4; ++jt)
            #pragma unroll
            for (int r = 0; r < 8; ++r)
                pbase[(r + half * 8) * LDT + jt * 16 + l16] = (f16)S[jt][r];

        // rescale O accumulators
        #pragma unroll
        for (int dt = 0; dt < 4; ++dt)
            #pragma unroll
            for (int r = 0; r < 8; ++r)
                o[dt][r] *= scale[r];

        // O += P @ V  (K-dim = j, 64 wide -> 2 WMMAs per d-tile)
        const f16* prow = &pbase[l16 * LDT];
        v16h pa0 = load_a_frag(prow, 0,  half);
        v16h pa1 = load_a_frag(prow, 32, half);
        #pragma unroll
        for (int dt = 0; dt < 4; ++dt) {
            const f16* vrow = &ldsV[cur][(dt * 16 + l16) * LDT];
            v16h bv0 = load_b_frag(vrow, 0,  half);
            v16h bv1 = load_b_frag(vrow, 32, half);
            o[dt] = wmma_f16(pa0, bv0, o[dt]);
            o[dt] = wmma_f16(pa1, bv1, o[dt]);
        }
        __syncthreads();
    }

    // epilogue: normalize and store f32 out[b, n, h*64 + d]
    const int trow0 = qb * 64 + wv * 16 + half * 8;
    #pragma unroll
    for (int dt = 0; dt < 4; ++dt) {
        int d = h * 64 + dt * 16 + l16;
        #pragma unroll
        for (int r = 0; r < 8; ++r) {
            int n = trow0 + r;
            out[((size_t)b * 1024 + n) * 768 + d] = o[dt][r] * (1.0f / lrow[r]);
        }
    }
}

// ---------------------------------------------------------------------------
// Host launcher. Workspace (f16):
//   xh  : 8192*768   = 6,291,456
//   whT : 2304*768   = 1,769,472
//   qh  : 6,291,456   kh : 6,291,456   vth : 6,291,456  (vT: [b,h,d,n])
// total ~53.9 MB.
// ---------------------------------------------------------------------------
extern "C" void kernel_launch(void* const* d_in, const int* in_sizes, int n_in,
                              void* d_out, int out_size, void* d_ws, size_t ws_size,
                              hipStream_t stream) {
    const float* x    = (const float*)d_in[0];
    const float* W    = (const float*)d_in[1];
    const float* bias = (const float*)d_in[2];
    float* out = (float*)d_out;

    f16* xh  = (f16*)d_ws;
    f16* whT = xh  + (size_t)6291456;
    f16* qh  = whT + (size_t)1769472;
    f16* kh  = qh  + (size_t)6291456;
    f16* vth = kh  + (size_t)6291456;

    convert_x <<<(8192 * 768) / 256, 256, 0, stream>>>(x, xh);
    convert_wT<<<(768 * 2304) / 256, 256, 0, stream>>>(W, whT);

    dim3 gGemm(128, 36);                      // 8192/64 x 2304/64
    qkv_gemm<<<gGemm, 128, 0, stream>>>(xh, whT, bias, qh, kh, vth);

    attn<<<96 * 16, 128, 0, stream>>>(qh, kh, vth, out);
}
